// gen_38800734552595
// MI455X (gfx1250) — matrix operand order repaired and checked by reading, not compiled, not run on hardware
//
#include <hip/hip_runtime.h>
#include <math.h>

#define H 128
#define N_NODES 100000
#define N_HE 20000
#define E_EDGES 640000
#define TAU 0.1f
#define LN_EPS 1e-5f
#define RT 4            // 16-row tiles per block in the GEMM

typedef __attribute__((ext_vector_type(16))) __bf16 v16bf;
typedef __attribute__((ext_vector_type(8)))  float  v8f;

union FragBF { v16bf v; unsigned int u[8]; uint4 q[2]; };
union FragF  { v8f v; float f[8]; };
union U4U    { uint4 d; unsigned short s[8]; };

__device__ __forceinline__ unsigned short f2bf(float f) {
  unsigned int u = __float_as_uint(f);
  u += 0x7FFFu + ((u >> 16) & 1u);   // round-to-nearest-even
  return (unsigned short)(u >> 16);
}

// ---------------- elementwise helpers ----------------

__global__ void fill_f32(float* __restrict__ p, float v, long long n) {
  long long i = (long long)blockIdx.x * blockDim.x + threadIdx.x;
  if (i < n) p[i] = v;
}

__global__ void cvt_bf16_k(const float* __restrict__ s, unsigned short* __restrict__ d, long long n) {
  long long i = (long long)blockIdx.x * blockDim.x + threadIdx.x;
  if (i < n) d[i] = f2bf(s[i]);
}

__global__ void softplus_k(float* __restrict__ p, long long n) {
  long long i = (long long)blockIdx.x * blockDim.x + threadIdx.x;
  if (i < n) { float x = p[i]; p[i] = (x > 20.f) ? x : log1pf(expf(x)); }
}

__global__ void reparam_k(const float* __restrict__ noise, const float* __restrict__ stdv,
                          const float* __restrict__ mean, float* __restrict__ z, long long n) {
  long long i = (long long)blockIdx.x * blockDim.x + threadIdx.x;
  if (i < n) z[i] = noise[i] * stdv[i] + mean[i];
}

// ---------------- LayerNorm (f32 in -> bf16 out), one 128-wide row per block ----------------

__global__ void ln_bf16_kernel(const float* __restrict__ x, const float* __restrict__ g,
                               const float* __restrict__ b, unsigned short* __restrict__ out) {
  __shared__ float s1[H], s2[H];
  int row = blockIdx.x, t = threadIdx.x;
  float v = x[(size_t)row * H + t];
  s1[t] = v; s2[t] = v * v;
  __syncthreads();
  for (int o = H / 2; o > 0; o >>= 1) {
    if (t < o) { s1[t] += s1[t + o]; s2[t] += s2[t + o]; }
    __syncthreads();
  }
  float mu  = s1[0] * (1.0f / H);
  float var = s2[0] * (1.0f / H) - mu * mu;
  float r = rsqrtf(var + LN_EPS);
  out[(size_t)row * H + t] = f2bf((v - mu) * r * g[t] + b[t]);
}

// ---------------- WMMA bf16 GEMM: out[M,128] = act(A[M,128] @ W[128,128] + bias) ----------------
// Block = 256 threads = 8 waves; block computes RT*16 rows x 128 cols; wave w -> cols [16w,16w+16).
// W staged once per block in LDS transposed [n][k]; B fragment shared across the RT row tiles;
// RT independent accumulator chains cover the WMMA->WMMA RAW hazard latency.

__global__ void __launch_bounds__(256) gemm_bf16_128(
    const unsigned short* __restrict__ A,    // [M,128] bf16 row-major
    const unsigned short* __restrict__ W,    // [128,128] bf16 row-major [k][n]
    const float* __restrict__ bias,          // [128]
    float* __restrict__ out,                 // [M,128] f32
    int M, int relu) {
  __shared__ unsigned short sA[RT * 16 * H];
  __shared__ unsigned short sBt[H * H];      // transposed: [n][k]
  __shared__ float sBias[H];

  int tid = threadIdx.x;
  size_t tile0 = (size_t)blockIdx.x * RT;    // in units of 16-row tiles

  // A tiles: each 16x128 bf16 = 256 x uint4 (one uint4 per thread per tile)
#pragma unroll
  for (int t = 0; t < RT; ++t) {
    size_t r0 = (tile0 + t) * 16;
    if (r0 < (size_t)M)
      ((uint4*)(sA + t * 16 * H))[tid] = ((const uint4*)(A + r0 * H))[tid];
  }
  // W: 2048 uint4, transposed into sBt[n][k]
#pragma unroll
  for (int it = 0; it < 8; ++it) {
    int lin = tid * 8 + it;                  // 0..2047
    int k  = lin >> 4;                       // 16 uint4 per 128-wide row
    int n0 = (lin & 15) << 3;
    U4U t4; t4.d = ((const uint4*)W)[lin];
#pragma unroll
    for (int q = 0; q < 8; ++q) sBt[(size_t)(n0 + q) * H + k] = t4.s[q];
  }
  if (tid < H) sBias[tid] = bias[tid];
  __syncthreads();

  int lane = tid & 31, wave = tid >> 5;
  int lm = lane & 15, kg = lane >> 4;
  int nG = wave * 16 + lm;

  // Fragment base pointers in uint4 units (16B = 8 bf16 = 4 dwords).
  // A frag (tile t, k-step kk): dwords lm*64 + kg*4 + kk/2 .. (+8 for upper half)
  //   -> uint4 idx: t*256 + lm*16 + kg + kk/8, second at +2
  // B frag: dwords nG*64 + kg*4 + kk/2 .. (+8 for upper half) -> uint4 idx nG*16 + kg + kk/8, second at +2
  const uint4* a4 = (const uint4*)sA + lm * 16 + kg;
  const uint4* b4 = (const uint4*)sBt + nG * 16 + kg;

  FragF c[RT];
#pragma unroll
  for (int t = 0; t < RT; ++t)
#pragma unroll
    for (int j = 0; j < 8; ++j) c[t].f[j] = 0.0f;

#pragma unroll
  for (int k8 = 0; k8 < 16; k8 += 4) {       // k8 = kk/8, kk in {0,32,64,96}
    FragBF bq;
    bq.q[0] = b4[k8];
    bq.q[1] = b4[k8 + 2];
    FragBF aq[RT];
#pragma unroll
    for (int t = 0; t < RT; ++t) {
      aq[t].q[0] = a4[t * 256 + k8];
      aq[t].q[1] = a4[t * 256 + k8 + 2];
    }
#pragma unroll
    for (int t = 0; t < RT; ++t)
      c[t].v = __builtin_amdgcn_wmma_f32_16x16x32_bf16(false, aq[t].v, false, bq.v,
                                                       (short)0, c[t].v, false, false);
  }

  float bv = sBias[nG];
#pragma unroll
  for (int t = 0; t < RT; ++t) {
    size_t r0 = (tile0 + t) * 16;
    if (r0 < (size_t)M) {
#pragma unroll
      for (int j = 0; j < 8; ++j) {
        int m = kg * 8 + j;                  // C/D: VGPR j -> M=j (lanes 0-15) / M=j+8 (lanes 16-31)
        float val = c[t].f[j] + bv;
        if (relu) val = fmaxf(val, 0.0f);
        out[(r0 + m) * H + nG] = val;
      }
    }
  }
}

// ---------------- message scatter: agg[sidx[e]] += norm[e] * h[gidx[e]] ----------------

__global__ void scatter_msg(const float* __restrict__ h, const int* __restrict__ gidx,
                            const int* __restrict__ sidx, const float* __restrict__ nrm,
                            float* __restrict__ agg) {
  long long i = (long long)blockIdx.x * blockDim.x + threadIdx.x;
  if (i >= (long long)E_EDGES * H) return;
  long long e = i >> 7; int hh = (int)(i & (H - 1));
  float m = nrm[e] * h[(size_t)gidx[e] * H + hh];
  atomicAdd(&agg[(size_t)sidx[e] * H + hh], m);
}

// ---------------- edge features: bf16(z_node[src] * z_he[dst]) ----------------

__global__ void edgefeat_k(const float* __restrict__ zn, const float* __restrict__ zh,
                           const int* __restrict__ src, const int* __restrict__ dst,
                           unsigned short* __restrict__ out) {
  long long i = (long long)blockIdx.x * blockDim.x + threadIdx.x;
  if (i >= (long long)E_EDGES * H) return;
  long long e = i >> 7; int hh = (int)(i & (H - 1));
  out[i] = f2bf(zn[(size_t)src[e] * H + hh] * zh[(size_t)dst[e] * H + hh]);
}

// ---------------- decoder head: logits[e,0:2] = h[e,:] @ W2 + b2 ----------------

__global__ void dec_head_k(const float* __restrict__ h, const float* __restrict__ W2,
                           const float* __restrict__ b2, float* __restrict__ logits) {
  long long e = (long long)blockIdx.x * blockDim.x + threadIdx.x;
  if (e >= E_EDGES) return;
  float a0 = b2[0], a1 = b2[1];
  const float* hp = h + (size_t)e * H;
#pragma unroll 4
  for (int k = 0; k < H; ++k) { float v = hp[k]; a0 += v * W2[2 * k]; a1 += v * W2[2 * k + 1]; }
  logits[2 * e] = a0; logits[2 * e + 1] = a1;
}

// ---------------- edge loss + gumbel softmax prediction ----------------

__global__ void edge_final_k(const float* __restrict__ logits, const float* __restrict__ ug,
                             float* __restrict__ pred, float* __restrict__ accLoss) {
  __shared__ float red[256];
  long long e = (long long)blockIdx.x * 256 + threadIdx.x;
  float contrib = 0.0f;
  if (e < E_EDGES) {
    float l0 = logits[2 * e], l1 = logits[2 * e + 1];
    float mx = fmaxf(l0, l1);
    float lse = mx + logf(expf(l0 - mx) + expf(l1 - mx));
    contrib = -(l1 - lse);                       // -log_softmax[:,1]
    float g0 = -logf(-logf(ug[2 * e]));
    float g1 = -logf(-logf(ug[2 * e + 1]));
    float t0 = (l0 + g0) * (1.0f / TAU), t1 = (l1 + g1) * (1.0f / TAU);
    float tm = fmaxf(t0, t1);
    float y0 = expf(t0 - tm), y1 = expf(t1 - tm);
    float s = y0 + y1; y0 /= s; y1 /= s;
    pred[e] = 1.0f / (1.0f + expf(y0 - y1));     // softmax(y+0.1)[:,1]; +0.1 cancels
  }
  red[threadIdx.x] = contrib;
  __syncthreads();
  for (int o = 128; o > 0; o >>= 1) {
    if (threadIdx.x < o) red[threadIdx.x] += red[threadIdx.x + o];
    __syncthreads();
  }
  if (threadIdx.x == 0) atomicAdd(accLoss, red[0]);
}

// ---------------- KL accumulation: sum(1 + 2 log std - mean^2 - std^2) ----------------

__global__ void kl_k(const float* __restrict__ mean, const float* __restrict__ stdv,
                     long long n, float* __restrict__ acc) {
  __shared__ float red[256];
  long long i = (long long)blockIdx.x * 256 + threadIdx.x;
  float t = 0.0f;
  if (i < n) { float m = mean[i], s = stdv[i]; t = 1.0f + 2.0f * logf(s) - m * m - s * s; }
  red[threadIdx.x] = t;
  __syncthreads();
  for (int o = 128; o > 0; o >>= 1) {
    if (threadIdx.x < o) red[threadIdx.x] += red[threadIdx.x + o];
    __syncthreads();
  }
  if (threadIdx.x == 0) atomicAdd(acc, red[0]);
}

__global__ void loss_final_k(const float* __restrict__ acc, float* __restrict__ outLoss) {
  float le = acc[0] / (float)E_EDGES;
  float kn = -0.5f * acc[1] / ((float)N_NODES * (float)N_NODES);
  float kh = -0.5f * acc[2] / ((float)N_HE * (float)N_HE);
  *outLoss = le + kn + kh;
}

// ---------------- host orchestration ----------------

static inline unsigned gblk(long long n, int b) { return (unsigned)((n + b - 1) / b); }

extern "C" void kernel_launch(void* const* d_in, const int* in_sizes, int n_in,
                              void* d_out, int out_size, void* d_ws, size_t ws_size,
                              hipStream_t stream) {
  (void)in_sizes; (void)n_in; (void)out_size; (void)ws_size;
  const float* x      = (const float*)d_in[0];
  const int*   hei    = (const int*)d_in[1];
  const int*   src    = hei;                // node ids
  const int*   dst    = hei + E_EDGES;      // hyperedge ids
  const float* nrm    = (const float*)d_in[2];
  const float* ln_g   = (const float*)d_in[3];   // [2,4,H]
  const float* ln_b   = (const float*)d_in[4];
  const float* convW  = (const float*)d_in[5];   // [2,4,H,H]
  const float* convB  = (const float*)d_in[6];   // [2,4,H]
  const float* encW1  = (const float*)d_in[7];   // [4,H,H]
  const float* encB1  = (const float*)d_in[8];
  const float* encW2  = (const float*)d_in[9];
  const float* encB2  = (const float*)d_in[10];
  const float* decW1  = (const float*)d_in[11];  // [H,H]
  const float* decB1  = (const float*)d_in[12];
  const float* decW2  = (const float*)d_in[13];  // [H,2]
  const float* decB2  = (const float*)d_in[14];
  const float* noiseN = (const float*)d_in[15];
  const float* noiseH = (const float*)d_in[16];
  const float* ug     = (const float*)d_in[17];
  float* outp = (float*)d_out;

  // workspace carve-up (256B aligned)
  size_t off = 0;
  char* wsb = (char*)d_ws;
  auto take = [&](size_t bytes) -> char* {
    char* p = wsb + off;
    off += (bytes + 255) & ~(size_t)255;
    return p;
  };
  const size_t MAT = (size_t)H * H;     // 16384
  float*          acc    = (float*)take(4 * sizeof(float));
  unsigned short* wBf    = (unsigned short*)take(17 * MAT * 2);     // bf16 weights
  unsigned short* bfBig  = (unsigned short*)take((size_t)E_EDGES * H * 2);
  float*          f32Big = (float*)take((size_t)E_EDGES * H * 4);
  float*          aggBuf = (float*)take((size_t)N_NODES * H * 4);
  float*          xHe    = (float*)take((size_t)N_HE * H * 4);
  float*          xNode  = (float*)take((size_t)N_NODES * H * 4);
  float*          meanN  = (float*)take((size_t)N_NODES * H * 4);
  float*          stdN   = (float*)take((size_t)N_NODES * H * 4);
  float*          zN     = (float*)take((size_t)N_NODES * H * 4);
  float*          meanHe = (float*)take((size_t)N_HE * H * 4);
  float*          stdHe  = (float*)take((size_t)N_HE * H * 4);
  float*          zHe    = (float*)take((size_t)N_HE * H * 4);
  float*          logits = (float*)take((size_t)E_EDGES * 2 * 4);

  // zero accumulators
  fill_f32<<<1, 32, 0, stream>>>(acc, 0.0f, 4);

  // convert all weight matrices to bf16: conv[0..7], encW1[8..11], encW2[12..15], decW1[16]
  cvt_bf16_k<<<gblk(8 * MAT, 256), 256, 0, stream>>>(convW, wBf + 0 * MAT, 8 * (long long)MAT);
  cvt_bf16_k<<<gblk(4 * MAT, 256), 256, 0, stream>>>(encW1, wBf + 8 * MAT, 4 * (long long)MAT);
  cvt_bf16_k<<<gblk(4 * MAT, 256), 256, 0, stream>>>(encW2, wBf + 12 * MAT, 4 * (long long)MAT);
  cvt_bf16_k<<<gblk(1 * MAT, 256), 256, 0, stream>>>(decW1, wBf + 16 * MAT, 1 * (long long)MAT);

  auto lnL = [&](const float* in, int c, int m, int rows) {
    ln_bf16_kernel<<<(unsigned)rows, H, 0, stream>>>(in, ln_g + (size_t)(c * 4 + m) * H,
                                                     ln_b + (size_t)(c * 4 + m) * H, bfBig);
  };
  auto gemm = [&](int wIdx, const float* bias, float* out_, int M, int relu) {
    unsigned blocks = (unsigned)((M / 16 + RT - 1) / RT);
    gemm_bf16_128<<<blocks, 256, 0, stream>>>(bfBig, wBf + (size_t)wIdx * MAT, bias, out_, M, relu);
  };

  // HalfNLHconv: relu(f_enc) -> scatter(norm * h[g]) -> relu(f_dec)
  auto halfconv = [&](const float* hin, int c, const int* gidx, const int* sidx,
                      int nin, int nout, float* outBuf) {
    lnL(hin, c, 0, nin);
    gemm(c * 4 + 0, convB + (size_t)(c * 4 + 0) * H, f32Big, nin, 1);
    lnL(f32Big, c, 1, nin);
    gemm(c * 4 + 1, convB + (size_t)(c * 4 + 1) * H, f32Big, nin, 1);   // relu(f_enc)
    fill_f32<<<gblk((long long)nout * H, 256), 256, 0, stream>>>(aggBuf, 0.0f, (long long)nout * H);
    scatter_msg<<<gblk((long long)E_EDGES * H, 256), 256, 0, stream>>>(f32Big, gidx, sidx, nrm, aggBuf);
    lnL(aggBuf, c, 2, nout);
    gemm(c * 4 + 2, convB + (size_t)(c * 4 + 2) * H, f32Big, nout, 1);
    lnL(f32Big, c, 3, nout);
    gemm(c * 4 + 3, convB + (size_t)(c * 4 + 3) * H, outBuf, nout, 1);  // relu(f_dec)
  };

  halfconv(x, 0, src, dst, N_NODES, N_HE, xHe);      // V2E
  halfconv(xHe, 1, dst, src, N_HE, N_NODES, xNode);  // E2V

  // VAE encoders: relu(h@W1+b1)@W2+b2
  auto encoder = [&](const float* hin, int idx, int rows, float* outBuf) {
    long long n = (long long)rows * H;
    cvt_bf16_k<<<gblk(n, 256), 256, 0, stream>>>(hin, bfBig, n);
    gemm(8 + idx, encB1 + (size_t)idx * H, f32Big, rows, 1);
    cvt_bf16_k<<<gblk(n, 256), 256, 0, stream>>>(f32Big, bfBig, n);
    gemm(12 + idx, encB2 + (size_t)idx * H, outBuf, rows, 0);
  };
  encoder(xNode, 0, N_NODES, meanN);
  encoder(xNode, 1, N_NODES, stdN);
  softplus_k<<<gblk((long long)N_NODES * H, 256), 256, 0, stream>>>(stdN, (long long)N_NODES * H);
  encoder(xHe, 2, N_HE, meanHe);
  encoder(xHe, 3, N_HE, stdHe);
  softplus_k<<<gblk((long long)N_HE * H, 256), 256, 0, stream>>>(stdHe, (long long)N_HE * H);

  reparam_k<<<gblk((long long)N_NODES * H, 256), 256, 0, stream>>>(noiseN, stdN, meanN, zN, (long long)N_NODES * H);
  reparam_k<<<gblk((long long)N_HE * H, 256), 256, 0, stream>>>(noiseH, stdHe, meanHe, zHe, (long long)N_HE * H);

  kl_k<<<gblk((long long)N_NODES * H, 256), 256, 0, stream>>>(meanN, stdN, (long long)N_NODES * H, acc + 1);
  kl_k<<<gblk((long long)N_HE * H, 256), 256, 0, stream>>>(meanHe, stdHe, (long long)N_HE * H, acc + 2);

  // edge decode
  edgefeat_k<<<gblk((long long)E_EDGES * H, 256), 256, 0, stream>>>(zN, zHe, src, dst, bfBig);
  gemm(16, decB1, f32Big, E_EDGES, 1);
  dec_head_k<<<gblk(E_EDGES, 256), 256, 0, stream>>>(f32Big, decW2, decB2, logits);
  edge_final_k<<<gblk(E_EDGES, 256), 256, 0, stream>>>(logits, ug, outp, acc + 0);

  loss_final_k<<<1, 1, 0, stream>>>(acc, outp + E_EDGES);
}
